// AlignmentModule_49950469652614
// MI455X (gfx1250) — matrix operand order repaired and checked by reading, not compiled, not run on hardware
//
#include <hip/hip_runtime.h>

// CDNA5 / gfx1250 WMMA implementation of the alignment module.
// All conv layers and the pairwise-distance dot products run through
// v_wmma_f32_16x16x32_f16 (wave32, 16x16x32, f16 inputs, f32 accum).
// GEMMs are register-blocked 1(M)x4(N) tiles per wave so each A fragment
// feeds 4 independent WMMA accumulators (less VMEM per FLOP, no D->A/B
// RAW hazard stalls between consecutive WMMAs).

typedef __attribute__((ext_vector_type(16))) _Float16 v16h;
typedef __attribute__((ext_vector_type(8)))  float    v8f;

namespace {
constexpr int kADIM = 256;
constexpr int kTT   = 160;   // T_text
constexpr int kTF   = 800;   // T_feats
constexpr int kB    = 4;
}

#define WMMA_F16(a, b, c) \
  __builtin_amdgcn_wmma_f32_16x16x32_f16(false, (a), false, (b), (short)0, (c), false, false)

// ---- ISA fragment layouts (CDNA5 ISA §7.12.2, wave32) ----
// 16-bit A matrix 16x32 (MxK): lane = M + 16*khalf; VGPR v holds 2 halves.
__device__ __forceinline__ int a_frag_k(int lane, int h) {
  int kh = lane >> 4, v = h >> 1, j = h & 1;
  return (v < 4) ? (2 * v + j + 8 * kh) : (16 + 2 * (v - 4) + j + 8 * kh);
}
// 16-bit B matrix 32x16 (KxN): lane = N + 16*khalf; same K order as A.
__device__ __forceinline__ int b_frag_k(int lane, int h) {
  int kh = lane >> 4, v = h >> 1, j = h & 1;
  return (v < 4) ? (2 * v + j + 8 * kh) : (16 + 2 * (v - 4) + j + 8 * kh);
}

// ---- pack conv weights (Cout, Cin, taps) f32 -> fragment-ready f16 A tiles ----
// Wp blocks: [mt][kc], each block = 32 lanes x 16 halves (lane-major).
__global__ void k_pack_w(const float* __restrict__ W, _Float16* __restrict__ Wp,
                         int Kreal, int numKc, int total) {
  int idx = blockIdx.x * blockDim.x + threadIdx.x;
  if (idx >= total) return;
  int h = idx & 15, lane = (idx >> 4) & 31, blk = idx >> 9;
  int kc = blk % numKc, mt = blk / numKc;
  int k  = kc * 32 + a_frag_k(lane, h);
  int co = mt * 16 + (lane & 15);
  float v = (k < Kreal) ? W[(size_t)co * Kreal + k] : 0.f;
  Wp[idx] = (_Float16)v;
}

// ---- pack im2col activations -> fragment-ready f16 B tiles ----
// X[b, ci, t] at b*sB + ci*sC + t*sT.  K-dim ordering = ci*taps + tap
// (matches (Cout,Cin,taps) weight memory order).  Xp blocks: [kc][nt].
__global__ void k_pack_x(const float* __restrict__ X, _Float16* __restrict__ Xp,
                         int T, int Cin, int taps, int pad,
                         int numNt, int sB, int sC, int sT, int total) {
  int idx = blockIdx.x * blockDim.x + threadIdx.x;
  if (idx >= total) return;
  int h = idx & 15, lane = (idx >> 4) & 31, blk = idx >> 9;
  int nt = blk % numNt, kc = blk / numNt;
  int n  = nt * 16 + (lane & 15);
  int k  = kc * 32 + b_frag_k(lane, h);
  int b  = n / T, t = n - b * T;
  int ci = k / taps, tap = k - ci * taps;
  int ts = t + tap - pad;
  float v = 0.f;
  if (ci < Cin && ts >= 0 && ts < T)
    v = X[(size_t)b * sB + (size_t)ci * sC + (size_t)ts * sT];
  Xp[idx] = (_Float16)v;
}

// ---- generic WMMA GEMM, 1x4 tile blocking per wave ----
// Y[b,co,t] = act(sum_k Wp*Xp + bias).  numNt must be a multiple of 4.
__global__ void k_gemm(const _Float16* __restrict__ Wp, const _Float16* __restrict__ Xp,
                       const float* __restrict__ bias, float* __restrict__ Y,
                       int numKc, int numNt, int numTiles4, int T, int Cdim, int relu) {
  int wave = threadIdx.x >> 5, lane = threadIdx.x & 31;
  int tile = blockIdx.x * (blockDim.x >> 5) + wave;
  if (tile >= numTiles4) return;                // wave-uniform: EXEC stays all-1s
  int numNt4 = numNt >> 2;
  int ntb = (tile % numNt4) << 2, mt = tile / numNt4;
  const v16h* A  = (const v16h*)Wp + (size_t)mt * numKc * 32;
  const v16h* Bm = (const v16h*)Xp;
  v8f acc[4] = {};
  for (int kc = 0; kc < numKc; ++kc) {
    v16h a = A[kc * 32 + lane];                 // A fragment reused 4x
    const v16h* bp = Bm + ((size_t)kc * numNt + ntb) * 32 + lane;
    v16h b0 = bp[0], b1 = bp[32], b2 = bp[64], b3 = bp[96];
    acc[0] = WMMA_F16(a, b0, acc[0]);
    acc[1] = WMMA_F16(a, b1, acc[1]);
    acc[2] = WMMA_F16(a, b2, acc[2]);
    acc[3] = WMMA_F16(a, b3, acc[3]);
  }
  int mbase = mt * 16 + 8 * (lane >> 4);        // C/D layout: VGPR r -> M = r (+8 hi lanes)
#pragma unroll
  for (int q = 0; q < 4; ++q) {
    int n = (ntb + q) * 16 + (lane & 15);
    int bb = n / T, t = n - bb * T;
#pragma unroll
    for (int r = 0; r < 8; ++r) {
      int co = mbase + r;
      float v = acc[q][r] + bias[co];
      if (relu) v = fmaxf(v, 0.f);
      Y[((size_t)bb * Cdim + co) * T + t] = v;
    }
  }
}

// ---- squared row norms over channels: out[b,t] = sum_c Y[b,c,t]^2 ----
__global__ void k_rownorm(const float* __restrict__ Y, float* __restrict__ out,
                          int C, int T, int total) {
  int idx = blockIdx.x * blockDim.x + threadIdx.x;
  if (idx >= total) return;
  int b = idx / T, t = idx - b * T;
  const float* p = Y + (size_t)b * C * T + t;
  float s = 0.f;
  for (int c = 0; c < C; ++c) { float v = p[(size_t)c * T]; s += v * v; }
  out[idx] = s;
}

// ---- pack feats output (B,256,800) as A fragments: rows = feat time i ----
__global__ void k_pack_af(const float* __restrict__ F, _Float16* __restrict__ Ap, int total) {
  int idx = blockIdx.x * blockDim.x + threadIdx.x;
  if (idx >= total) return;
  int h = idx & 15, lane = (idx >> 4) & 31, blk = idx >> 9;  // [b][mt(50)][kc(8)]
  int kc = blk % 8, mt = (blk / 8) % (kTF / 16), b = blk / (8 * (kTF / 16));
  int i = mt * 16 + (lane & 15);
  int k = kc * 32 + a_frag_k(lane, h);
  Ap[idx] = (_Float16)F[((size_t)b * kADIM + k) * kTF + i];
}

// ---- pack text output (B,256,160) as B fragments: cols = text time j ----
__global__ void k_pack_bt(const float* __restrict__ Tm, _Float16* __restrict__ Bp, int total) {
  int idx = blockIdx.x * blockDim.x + threadIdx.x;
  if (idx >= total) return;
  int h = idx & 15, lane = (idx >> 4) & 31, blk = idx >> 9;  // [b][kc(8)][nt(10)]
  int nt = blk % (kTT / 16), kc = (blk / (kTT / 16)) % 8, b = blk / (8 * (kTT / 16));
  int j = nt * 16 + (lane & 15);
  int k = kc * 32 + b_frag_k(lane, h);
  Bp[idx] = (_Float16)Tm[((size_t)b * kADIM + k) * kTT + j];
}

// ---- distance GEMM (1x2 N-blocked): score[b,i,j] = -sqrt(nf+nt-2*dot) ----
__global__ void k_dist(const _Float16* __restrict__ Ap, const _Float16* __restrict__ Bp,
                       const float* __restrict__ nf, const float* __restrict__ nt2,
                       const unsigned char* __restrict__ mask, float* __restrict__ S) {
  constexpr int numMt = kTF / 16, numNt = kTT / 16, numKc = kADIM / 32;
  constexpr int numNt2 = numNt / 2;
  int wave = threadIdx.x >> 5, lane = threadIdx.x & 31;
  int tile = blockIdx.x * (blockDim.x >> 5) + wave;
  if (tile >= kB * numMt * numNt2) return;
  int ntb = (tile % numNt2) * 2;
  int mt = (tile / numNt2) % numMt, b = tile / (numNt2 * numMt);
  const v16h* A  = (const v16h*)Ap + ((size_t)(b * numMt + mt) * numKc) * 32;
  const v16h* Bm = (const v16h*)Bp + ((size_t)b * numKc * numNt) * 32;
  v8f acc[2] = {};
#pragma unroll
  for (int kc = 0; kc < numKc; ++kc) {
    v16h a = A[kc * 32 + lane];                 // A fragment reused 2x
    const v16h* bp = Bm + ((size_t)kc * numNt + ntb) * 32 + lane;
    v16h b0 = bp[0], b1 = bp[32];
    acc[0] = WMMA_F16(a, b0, acc[0]);
    acc[1] = WMMA_F16(a, b1, acc[1]);
  }
  int ibase = mt * 16 + 8 * (lane >> 4);
#pragma unroll
  for (int q = 0; q < 2; ++q) {
    int j = (ntb + q) * 16 + (lane & 15);
    bool msk = mask[b * kTT + j] != 0;
    float ntj = nt2[b * kTT + j];
#pragma unroll
    for (int r = 0; r < 8; ++r) {
      int i = ibase + r;
      float d2 = nf[b * kTF + i] + ntj - 2.f * acc[q][r];
      float sc = -sqrtf(fmaxf(d2, 0.f));
      if (msk) sc = -__builtin_inff();
      S[((size_t)(b * kTF + i)) * kTT + j] = sc;
    }
  }
}

// ---- beta-binomial prior (T=800, N=160), matches reference formula ----
__global__ void k_prior(float* __restrict__ P, int total) {
  int idx = blockIdx.x * blockDim.x + threadIdx.x;
  if (idx >= total) return;
  int i = idx / kTT, j = idx - (idx / kTT) * kTT;
  float a  = (float)(i + 1);           // w * arange(1, T+1)
  float b  = (float)(kTF - i);         // w * (T - a + 1)
  float k  = (float)j;
  float Nf = (float)kTT;
  float lp = lgammaf(Nf + 1.f) - lgammaf(k + 1.f) - lgammaf(Nf - k + 1.f)
           + lgammaf(k + a) + lgammaf(Nf - k + b) - lgammaf(Nf + a + b)
           - (lgammaf(a) + lgammaf(b) - lgammaf(a + b));
  P[idx] = lp;
}

// ---- log-softmax over T_text + add prior; one wave32 per (b,i) row ----
__global__ void k_softmax_prior(const float* __restrict__ S, const float* __restrict__ P,
                                float* __restrict__ out, int rows) {
  int wave = threadIdx.x >> 5, lane = threadIdx.x & 31;
  int row = blockIdx.x * (blockDim.x >> 5) + wave;
  if (row >= rows) return;
  int i = row % kTF;
  const float* g = S + (size_t)row * kTT;
  float mx = -3.4e38f;
  for (int j = lane; j < kTT; j += 32) mx = fmaxf(mx, g[j]);
  for (int o = 16; o > 0; o >>= 1) mx = fmaxf(mx, __shfl_xor(mx, o, 32));
  float sum = 0.f;
  for (int j = lane; j < kTT; j += 32) sum += expf(g[j] - mx);
  for (int o = 16; o > 0; o >>= 1) sum += __shfl_xor(sum, o, 32);
  float lse = mx + logf(sum);
  for (int j = lane; j < kTT; j += 32)
    out[(size_t)row * kTT + j] = g[j] - lse + P[i * kTT + j];
}

extern "C" void kernel_launch(void* const* d_in, const int* in_sizes, int n_in,
                              void* d_out, int out_size, void* d_ws, size_t ws_size,
                              hipStream_t stream) {
  (void)in_sizes; (void)n_in; (void)out_size; (void)ws_size;
  const float* text  = (const float*)d_in[0];   // (4,160,256)
  const float* feats = (const float*)d_in[1];   // (4,800,80)
  const unsigned char* xmask = (const unsigned char*)d_in[4];  // (4,160) bool
  const float* t_w1 = (const float*)d_in[5];
  const float* t_b1 = (const float*)d_in[6];
  const float* t_w2 = (const float*)d_in[7];
  const float* t_b2 = (const float*)d_in[8];
  const float* f_w1 = (const float*)d_in[9];
  const float* f_b1 = (const float*)d_in[10];
  const float* f_w2 = (const float*)d_in[11];
  const float* f_b2 = (const float*)d_in[12];
  const float* f_w3 = (const float*)d_in[13];
  const float* f_b3 = (const float*)d_in[14];
  float* out = (float*)d_out;

  unsigned char* ws = (unsigned char*)d_ws;
  size_t off = 0;
  auto alloc = [&](size_t bytes) -> void* {
    void* p = ws + off;
    off = (off + bytes + 255) & ~(size_t)255;
    return p;
  };

  // fragment half counts
  const int WP1T = 16 * 24 * 512, WP2T = 16 * 8 * 512;
  const int WP1F = 16 * 8 * 512,  WP2F = 16 * 24 * 512, WP3F = 16 * 8 * 512;
  const int XPMAX = 24 * 200 * 512;
  const int APF = kB * (kTF / 16) * 8 * 512;
  const int BPT = kB * 8 * (kTT / 16) * 512;

  _Float16* Wp1t = (_Float16*)alloc((size_t)WP1T * 2);
  _Float16* Wp2t = (_Float16*)alloc((size_t)WP2T * 2);
  _Float16* Wp1f = (_Float16*)alloc((size_t)WP1F * 2);
  _Float16* Wp2f = (_Float16*)alloc((size_t)WP2F * 2);
  _Float16* Wp3f = (_Float16*)alloc((size_t)WP3F * 2);
  _Float16* Xp   = (_Float16*)alloc((size_t)XPMAX * 2);
  _Float16* ApF  = (_Float16*)alloc((size_t)APF * 2);
  _Float16* BpT  = (_Float16*)alloc((size_t)BPT * 2);
  float* Yt1 = (float*)alloc((size_t)kB * kADIM * kTT * 4);
  float* Yt2 = (float*)alloc((size_t)kB * kADIM * kTT * 4);
  float* Yf1 = (float*)alloc((size_t)kB * kADIM * kTF * 4);
  float* Yf2 = (float*)alloc((size_t)kB * kADIM * kTF * 4);
  float* Yf3 = (float*)alloc((size_t)kB * kADIM * kTF * 4);
  float* nf  = (float*)alloc((size_t)kB * kTF * 4);
  float* nt2 = (float*)alloc((size_t)kB * kTT * 4);
  float* Pr  = (float*)alloc((size_t)kTF * kTT * 4);
  float* Sc  = (float*)alloc((size_t)kB * kTF * kTT * 4);

  const int TPB = 256;
  auto nb = [](int n) { return (n + 255) / 256; };

  // prior (independent of data)
  k_prior<<<nb(kTF * kTT), TPB, 0, stream>>>(Pr, kTF * kTT);

  // weight packing
  k_pack_w<<<nb(WP1T), TPB, 0, stream>>>(t_w1, Wp1t, 768, 24, WP1T);
  k_pack_w<<<nb(WP2T), TPB, 0, stream>>>(t_w2, Wp2t, 256, 8,  WP2T);
  k_pack_w<<<nb(WP1F), TPB, 0, stream>>>(f_w1, Wp1f, 240, 8,  WP1F);
  k_pack_w<<<nb(WP2F), TPB, 0, stream>>>(f_w2, Wp2f, 768, 24, WP2F);
  k_pack_w<<<nb(WP3F), TPB, 0, stream>>>(f_w3, Wp3f, 256, 8,  WP3F);

  // ---- text stem: conv(k3,relu) -> conv(k1) ----
  {
    int tot = 24 * 40 * 512;  // numKc=24, numNt=40 (N = 4*160)
    k_pack_x<<<nb(tot), TPB, 0, stream>>>(text, Xp, kTT, 256, 3, 1, 40,
                                          kTT * kADIM, 1, kADIM, tot);
    int tiles4 = 16 * (40 / 4);
    k_gemm<<<(tiles4 + 7) / 8, TPB, 0, stream>>>(Wp1t, Xp, t_b1, Yt1, 24, 40, tiles4, kTT, kADIM, 1);
    tot = 8 * 40 * 512;
    k_pack_x<<<nb(tot), TPB, 0, stream>>>(Yt1, Xp, kTT, 256, 1, 0, 40,
                                          kADIM * kTT, kTT, 1, tot);
    k_gemm<<<(tiles4 + 7) / 8, TPB, 0, stream>>>(Wp2t, Xp, t_b2, Yt2, 8, 40, tiles4, kTT, kADIM, 0);
  }

  // ---- feats stem: conv(k3,relu) -> conv(k3,relu) -> conv(k1) ----
  {
    int tot = 8 * 200 * 512;  // K=240 padded to 256, N = 4*800
    k_pack_x<<<nb(tot), TPB, 0, stream>>>(feats, Xp, kTF, 80, 3, 1, 200,
                                          kTF * 80, 1, 80, tot);
    int tiles4 = 16 * (200 / 4);
    k_gemm<<<(tiles4 + 7) / 8, TPB, 0, stream>>>(Wp1f, Xp, f_b1, Yf1, 8, 200, tiles4, kTF, kADIM, 1);
    tot = 24 * 200 * 512;
    k_pack_x<<<nb(tot), TPB, 0, stream>>>(Yf1, Xp, kTF, 256, 3, 1, 200,
                                          kADIM * kTF, kTF, 1, tot);
    k_gemm<<<(tiles4 + 7) / 8, TPB, 0, stream>>>(Wp2f, Xp, f_b2, Yf2, 24, 200, tiles4, kTF, kADIM, 1);
    tot = 8 * 200 * 512;
    k_pack_x<<<nb(tot), TPB, 0, stream>>>(Yf2, Xp, kTF, 256, 1, 0, 200,
                                          kADIM * kTF, kTF, 1, tot);
    k_gemm<<<(tiles4 + 7) / 8, TPB, 0, stream>>>(Wp3f, Xp, f_b3, Yf3, 8, 200, tiles4, kTF, kADIM, 0);
  }

  // ---- norms, distance GEMM, softmax + prior ----
  k_rownorm<<<nb(kB * kTF), TPB, 0, stream>>>(Yf3, nf,  kADIM, kTF, kB * kTF);
  k_rownorm<<<nb(kB * kTT), TPB, 0, stream>>>(Yt2, nt2, kADIM, kTT, kB * kTT);
  k_pack_af<<<nb(APF), TPB, 0, stream>>>(Yf3, ApF, APF);
  k_pack_bt<<<nb(BPT), TPB, 0, stream>>>(Yt2, BpT, BPT);
  int dtiles = kB * (kTF / 16) * ((kTT / 16) / 2);
  k_dist<<<(dtiles + 7) / 8, TPB, 0, stream>>>(ApF, BpT, nf, nt2, xmask, Sc);
  int rows = kB * kTF;
  k_softmax_prior<<<(rows + 7) / 8, TPB, 0, stream>>>(Sc, Pr, out, rows);
}
